// CCAM_13623636263266
// MI455X (gfx1250) — matrix operand order repaired and checked by reading, not compiled, not run on hardware
//
#include <hip/hip_runtime.h>

typedef unsigned short u16;
typedef unsigned int u32;

typedef __attribute__((ext_vector_type(16))) __bf16 v16bf;
typedef __attribute__((ext_vector_type(8)))  float  v8f;

#if defined(__HIP_DEVICE_COMPILE__)
#if __has_builtin(__builtin_amdgcn_global_load_async_to_lds_b128) && __has_builtin(__builtin_amdgcn_s_wait_asynccnt)
#define CDNA5_ASYNC 1
#endif
#endif
#ifndef CDNA5_ASYNC
#define CDNA5_ASYNC 0
#endif

#if CDNA5_ASYNC
// Matches the builtin's expected parameter types (from hipcc diagnostics):
//   arg0: AS1 (global, prints as __device__) pointer to 'int __vector(4)'
//   arg1: AS3 (LDS) pointer to 'int __vector(4)'
//   arg2/arg3: immediate offset / cpol
typedef int v4i_ls __attribute__((__vector_size__(4 * sizeof(int))));
typedef __attribute__((address_space(1))) v4i_ls* gptr_v4i;
typedef __attribute__((address_space(3))) v4i_ls* lptr_v4i;
#define ASYNC_CP16(gp, lp, off) \
  __builtin_amdgcn_global_load_async_to_lds_b128((gptr_v4i)(gp), (lptr_v4i)(lp), (off), 0)
#define ASYNC_WAIT() __builtin_amdgcn_s_wait_asynccnt(0)
#else
#define ASYNC_WAIT()
#endif

// ---------- helpers ----------
__device__ inline u16 f2bf(float f) {
  u32 u = __float_as_uint(f);
  u32 r = u + 0x7FFFu + ((u >> 16) & 1u);   // round-to-nearest-even
  return (u16)(r >> 16);
}
__device__ inline float bf2f(u16 h) { return __uint_as_float(((u32)h) << 16); }

__device__ inline float blockReduceSum(float v) {
  __shared__ float sm[8];
  #pragma unroll
  for (int o = 16; o; o >>= 1) v += __shfl_xor(v, o, 32);
  int w = threadIdx.x >> 5;
  if ((threadIdx.x & 31) == 0) sm[w] = v;
  __syncthreads();
  float r = sm[0];
  #pragma unroll
  for (int i = 1; i < 8; i++) r += sm[i];
  __syncthreads();
  return r;
}
__device__ inline float blockReduceMax(float v) {
  __shared__ float sm[8];
  #pragma unroll
  for (int o = 16; o; o >>= 1) v = fmaxf(v, __shfl_xor(v, o, 32));
  int w = threadIdx.x >> 5;
  if ((threadIdx.x & 31) == 0) sm[w] = v;
  __syncthreads();
  float r = sm[0];
  #pragma unroll
  for (int i = 1; i < 8; i++) r = fmaxf(r, sm[i]);
  __syncthreads();
  return r;
}

// ---------- K1: avgpool + bf16 cast + row mean ----------
__global__ __launch_bounds__(256) void prep_kernel(
    const float* __restrict__ x, u16* __restrict__ Abf,
    u16* __restrict__ Fbf, float* __restrict__ meanA) {
  const int bc = blockIdx.x;               // b*512 + c
  const int tid = threadIdx.x;
  const float* xb = x + (size_t)bc * 4096;

  // cast row to bf16 + accumulate sum (16 floats / thread)
  const float4* xr = (const float4*)xb;
  uint2* ar = (uint2*)(Abf + (size_t)bc * 4096);
  float sum = 0.f;
  #pragma unroll
  for (int t = 0; t < 4; t++) {
    float4 v = xr[tid + 256 * t];
    sum += v.x + v.y + v.z + v.w;
    uint2 p;
    p.x = (u32)f2bf(v.x) | ((u32)f2bf(v.y) << 16);
    p.y = (u32)f2bf(v.z) | ((u32)f2bf(v.w) << 16);
    ar[tid + 256 * t] = p;
  }
  float tot = blockReduceSum(sum);
  if (tid == 0) meanA[bc] = tot * (1.f / 4096.f);

  // 2x2 average pool: 4 consecutive pooled pixels / thread
  int p0 = tid * 4;
  int i = p0 >> 5, j0 = p0 & 31;
  const float* r0 = xb + (2 * i) * 64 + 2 * j0;
  float4 a0 = ((const float4*)r0)[0], a1 = ((const float4*)r0)[1];
  float4 b0 = ((const float4*)(r0 + 64))[0], b1 = ((const float4*)(r0 + 64))[1];
  float p0v = (a0.x + a0.y + b0.x + b0.y) * 0.25f;
  float p1v = (a0.z + a0.w + b0.z + b0.w) * 0.25f;
  float p2v = (a1.x + a1.y + b1.x + b1.y) * 0.25f;
  float p3v = (a1.z + a1.w + b1.z + b1.w) * 0.25f;
  uint2 pk;
  pk.x = (u32)f2bf(p0v) | ((u32)f2bf(p1v) << 16);
  pk.y = (u32)f2bf(p2v) | ((u32)f2bf(p3v) << 16);
  ((uint2*)(Fbf + (size_t)bc * 1024))[tid] = pk;
}

// ---------- K3: row softmax (in place) + bf16 copy ----------
__global__ __launch_bounds__(256) void softmax_kernel(
    float* __restrict__ G, u16* __restrict__ ScBf) {
  const size_t row = blockIdx.x;           // b*512 + c
  float* g = G + row * 512;
  const int tid = threadIdx.x;
  float v0 = g[tid], v1 = g[tid + 256];
  float m = blockReduceMax(fmaxf(v0, v1));
  float e0 = __expf(v0 - m), e1 = __expf(v1 - m);
  float s = blockReduceSum(e0 + e1);
  float inv = 1.f / s;
  e0 *= inv; e1 *= inv;
  g[tid] = e0; g[tid + 256] = e1;
  u16* sb = ScBf + row * 512;
  sb[tid] = f2bf(e0); sb[tid + 256] = f2bf(e1);
}

// ---------- K5: bilinear 2x upsample (half-pixel) + mean ----------
__global__ __launch_bounds__(256) void upsample_kernel(
    const u16* __restrict__ Mcb, u16* __restrict__ Mcup, float* __restrict__ meanM) {
  const int bc = blockIdx.x;
  const int tid = threadIdx.x;
  __shared__ float s[1024];
  uint2 v = ((const uint2*)(Mcb + (size_t)bc * 1024))[tid];
  s[tid * 4 + 0] = bf2f((u16)(v.x & 0xFFFF));
  s[tid * 4 + 1] = bf2f((u16)(v.x >> 16));
  s[tid * 4 + 2] = bf2f((u16)(v.y & 0xFFFF));
  s[tid * 4 + 3] = bf2f((u16)(v.y >> 16));
  __syncthreads();
  u16* dst = Mcup + (size_t)bc * 4096;
  float sum = 0.f;
  #pragma unroll
  for (int t = 0; t < 16; t++) {
    int q = tid + 256 * t;
    int i = q >> 6, j = q & 63;
    int pi = i >> 1, pj = j >> 1;
    int i0, i1, j0, j1; float wi0, wj0;
    if (i & 1) { i0 = pi; i1 = (pi + 1 < 32) ? pi + 1 : 31; wi0 = 0.75f; }
    else       { i0 = (pi > 0) ? pi - 1 : 0; i1 = pi; wi0 = 0.25f; }
    if (j & 1) { j0 = pj; j1 = (pj + 1 < 32) ? pj + 1 : 31; wj0 = 0.75f; }
    else       { j0 = (pj > 0) ? pj - 1 : 0; j1 = pj; wj0 = 0.25f; }
    float wi1 = 1.f - wi0, wj1 = 1.f - wj0;
    float val = wj0 * (wi0 * s[i0 * 32 + j0] + wi1 * s[i1 * 32 + j0])
              + wj1 * (wi0 * s[i0 * 32 + j1] + wi1 * s[i1 * 32 + j1]);
    sum += val;
    dst[q] = f2bf(val);
  }
  float tot = blockReduceSum(sum);
  if (tid == 0) meanM[bc] = tot * (1.f / 4096.f);
}

// ---------- WMMA bf16 GEMM: D = A * op(B), 128x128x32 tiles ----------
// BNT=true : B source is [N,K] row-major (computes A * B^T), direct LDS copy
// BNT=false: B source is [K,N] row-major (computes A * B), transpose via regs
// MODE 0: C0 = f32 result (raw gram)
// MODE 1: C1 = bf16 result (Mc)
// MODE 2: cov = acc/4096 - meanA[r]*meanB[c]; C0=cov f32; C1=bf16(cov+Sc)
// MODE 3: Ec = acc; C0=Ec f32; Out = X*(beta*Ec + X)
#define BM 128
#define BN 128
#define BKK 32
#define PITCH 40   // u16 per LDS row (80B, bank-conflict-free B128 frag loads)

template <int MODE, bool BNT>
__global__ __launch_bounds__(256) void gemm_bf16_kernel(
    const u16* __restrict__ Abase, int lda, size_t sA,
    const u16* __restrict__ Bbase, int ldb, size_t sB,
    int K,
    float* __restrict__ C0, u16* __restrict__ C1, int ldc, size_t sC,
    const float* __restrict__ meanA, const float* __restrict__ meanB,
    const float* __restrict__ ScBase,
    const float* __restrict__ Xbase, const float* __restrict__ betaP,
    float* __restrict__ OutBase) {
  __shared__ __align__(16) u16 lA[2][BM * PITCH];
  __shared__ __align__(16) u16 lB[2][BN * PITCH];

  const int bz = blockIdx.z;
  const u16* A  = Abase + (size_t)bz * sA;
  const u16* Bm = Bbase + (size_t)bz * sB;

  const int tid = threadIdx.x;
  const int lane = tid & 31;
  const int lrow = lane & 15, lhalf = lane >> 4;
  const int wave = tid >> 5;
  const int wrow = (wave >> 2) * 64;   // wave M offset in tile
  const int wcol = (wave & 3) * 32;    // wave N offset in tile

  const int rowBase = blockIdx.y * BM;
  const int colBase = blockIdx.x * BN;

  // cooperative-copy thread mapping
  const int crow = tid >> 1, chalf = tid & 1;     // contiguous copies
  const int kp = tid & 15, nc = (tid >> 4) * 8;   // NN transpose mapping

  const int KT = K / BKK;
  v8f acc[4][2] = {};
  uint4 rA0{}, rA1{}, rB0{}, rB1{};

  auto stage_prefetch = [&](int kt, int buf) {
    {
      const u16* ga = A + (size_t)(rowBase + crow) * lda + (size_t)kt * BKK + chalf * 16;
#if CDNA5_ASYNC
      u16* la = &lA[buf][crow * PITCH + chalf * 16];
      ASYNC_CP16(ga, la, 0);
      ASYNC_CP16(ga, la, 16);
#else
      rA0 = *(const uint4*)ga;
      rA1 = *(const uint4*)(ga + 8);
#endif
    }
    if (BNT) {
      const u16* gb = Bm + (size_t)(colBase + crow) * ldb + (size_t)kt * BKK + chalf * 16;
#if CDNA5_ASYNC
      u16* lb = &lB[buf][crow * PITCH + chalf * 16];
      ASYNC_CP16(gb, lb, 0);
      ASYNC_CP16(gb, lb, 16);
#else
      rB0 = *(const uint4*)gb;
      rB1 = *(const uint4*)(gb + 8);
#endif
    } else {
      const u16* gb = Bm + ((size_t)kt * BKK + 2 * kp) * ldb + colBase + nc;
      rB0 = *(const uint4*)gb;
      rB1 = *(const uint4*)(gb + ldb);
    }
    (void)buf;
  };

  auto stage_commit = [&](int buf) {
#if !CDNA5_ASYNC
    {
      u16* la = &lA[buf][crow * PITCH + chalf * 16];
      *(uint4*)la = rA0;
      *(uint4*)(la + 8) = rA1;
    }
    if (BNT) {
      u16* lb = &lB[buf][crow * PITCH + chalf * 16];
      *(uint4*)lb = rB0;
      *(uint4*)(lb + 8) = rB1;
    }
#endif
    if (!BNT) {  // pack k/k+1 pairs -> conflict-free ds_store_b32 transpose
      u32* lb32 = (u32*)&lB[buf][0];
      const u16* q0 = (const u16*)&rB0;
      const u16* q1 = (const u16*)&rB1;
      #pragma unroll
      for (int e = 0; e < 8; e++)
        lb32[(nc + e) * 20 + kp] = (u32)q0[e] | ((u32)q1[e] << 16);
    }
    (void)buf;
  };

  union FragU { uint4 q[2]; v16bf v; };

  stage_prefetch(0, 0);
  stage_commit(0);
  ASYNC_WAIT();
  __syncthreads();

  for (int kt = 0; kt < KT; ++kt) {
    const int cur = kt & 1;
    if (kt + 1 < KT) stage_prefetch(kt + 1, cur ^ 1);

    FragU fa[4], fb[2];
    #pragma unroll
    for (int i = 0; i < 4; i++) {
      const u16* pa = &lA[cur][(wrow + i * 16 + lrow) * PITCH + lhalf * 8];
      fa[i].q[0] = *(const uint4*)pa;          // K 0..7 (+8*lhalf)
      fa[i].q[1] = *(const uint4*)(pa + 16);   // K 16..23 (+8*lhalf)
    }
    #pragma unroll
    for (int j = 0; j < 2; j++) {
      const u16* pb = &lB[cur][(wcol + j * 16 + lrow) * PITCH + lhalf * 8];
      fb[j].q[0] = *(const uint4*)pb;          // K 0..7 (+8*lhalf)
      fb[j].q[1] = *(const uint4*)(pb + 16);   // K 16..23 (+8*lhalf)
    }
    #pragma unroll
    for (int i = 0; i < 4; i++)
      #pragma unroll
      for (int j = 0; j < 2; j++)
        acc[i][j] = __builtin_amdgcn_wmma_f32_16x16x32_bf16(
            false, fa[i].v, false, fb[j].v, (short)0, acc[i][j], false, false);

    if (kt + 1 < KT) stage_commit(cur ^ 1);
    ASYNC_WAIT();
    __syncthreads();
  }

  // epilogue
  const size_t cOff = (size_t)bz * sC;
  const float bval = (MODE == 3) ? betaP[0] : 0.0f;
  #pragma unroll
  for (int i = 0; i < 4; i++) {
    #pragma unroll
    for (int j = 0; j < 2; j++) {
      const int r0 = rowBase + wrow + i * 16 + lhalf * 8;
      const int c0 = colBase + wcol + j * 16 + lrow;
      const float* ap = (const float*)&acc[i][j];
      #pragma unroll
      for (int v = 0; v < 8; v++) {
        const int r = r0 + v;
        const size_t idx = cOff + (size_t)r * ldc + c0;
        const float val = ap[v];
        if (MODE == 0) {
          C0[idx] = val;
        } else if (MODE == 1) {
          C1[idx] = f2bf(val);
        } else if (MODE == 2) {
          float cv = val * (1.f / 4096.f) - meanA[bz * 512 + r] * meanB[bz * 512 + c0];
          C0[idx] = cv;
          C1[idx] = f2bf(cv + ScBase[idx]);
        } else {
          float xv = Xbase[idx];
          C0[idx] = val;                              // Ec
          OutBase[idx] = xv * (bval * val + xv);      // out
        }
      }
    }
  }
}

// ---------- host launcher ----------
extern "C" void kernel_launch(void* const* d_in, const int* in_sizes, int n_in,
                              void* d_out, int out_size, void* d_ws, size_t ws_size,
                              hipStream_t stream) {
  (void)in_sizes; (void)n_in; (void)out_size; (void)ws_size;
  const float* x    = (const float*)d_in[0];
  const float* beta = (const float*)d_in[1];

  const int Bn = 16, Cn = 512;
  const size_t HW = 4096, NP = 1024;

  float* O    = (float*)d_out;
  float* outP = O;
  float* covP = O + (size_t)Bn * Cn * HW;
  float* scP  = covP + (size_t)Bn * Cn * Cn;
  float* ecP  = scP + (size_t)Bn * Cn * Cn;

  char* w = (char*)d_ws;
  auto take = [&](size_t bytes) -> char* {
    char* p = w; w += (bytes + 255) & ~(size_t)255; return p;
  };
  u16* Abf   = (u16*)take((size_t)Bn * Cn * HW * 2);
  u16* Fbf   = (u16*)take((size_t)Bn * Cn * NP * 2);
  u16* ScBf  = (u16*)take((size_t)Bn * Cn * Cn * 2);
  u16* Mcb   = (u16*)take((size_t)Bn * Cn * NP * 2);
  u16* Mcup  = (u16*)take((size_t)Bn * Cn * HW * 2);
  u16* LcBf  = (u16*)take((size_t)Bn * Cn * Cn * 2);
  float* mA  = (float*)take((size_t)Bn * Cn * 4);
  float* mM  = (float*)take((size_t)Bn * Cn * 4);

  // 1) pool + bf16 cast + row means
  prep_kernel<<<dim3(Bn * Cn), 256, 0, stream>>>(x, Abf, Fbf, mA);

  // 2) G = F * F^T  -> raw into Sc output region
  gemm_bf16_kernel<0, true><<<dim3(4, 4, Bn), 256, 0, stream>>>(
      Fbf, 1024, (size_t)Cn * NP, Fbf, 1024, (size_t)Cn * NP, 1024,
      scP, nullptr, 512, (size_t)Cn * Cn,
      nullptr, nullptr, nullptr, nullptr, nullptr, nullptr);

  // 3) softmax rows (in place) + bf16 copy
  softmax_kernel<<<dim3(Bn * Cn), 256, 0, stream>>>(scP, ScBf);

  // 4) Mc = Sc * F   (bf16 out)
  gemm_bf16_kernel<1, false><<<dim3(8, 4, Bn), 256, 0, stream>>>(
      ScBf, 512, (size_t)Cn * Cn, Fbf, 1024, (size_t)Cn * NP, 512,
      nullptr, Mcb, 1024, (size_t)Cn * NP,
      nullptr, nullptr, nullptr, nullptr, nullptr, nullptr);

  // 5) bilinear 2x upsample + meanM
  upsample_kernel<<<dim3(Bn * Cn), 256, 0, stream>>>(Mcb, Mcup, mM);

  // 6) cov = A*Mcup^T/hw - meanA x meanM ; Lc = bf16(cov + Sc)
  gemm_bf16_kernel<2, true><<<dim3(4, 4, Bn), 256, 0, stream>>>(
      Abf, 4096, (size_t)Cn * HW, Mcup, 4096, (size_t)Cn * HW, 4096,
      covP, LcBf, 512, (size_t)Cn * Cn,
      mA, mM, scP, nullptr, nullptr, nullptr);

  // 7) Ec = Lc * A ; out = x*(beta*Ec + x)
  gemm_bf16_kernel<3, false><<<dim3(32, 4, Bn), 256, 0, stream>>>(
      LcBf, 512, (size_t)Cn * Cn, Abf, 4096, (size_t)Cn * HW, 512,
      ecP, nullptr, 4096, (size_t)Cn * HW,
      nullptr, nullptr, nullptr, x, beta, outP);
}
